// CViViT_2293512536410
// MI455X (gfx1250) — matrix operand order repaired and checked by reading, not compiled, not run on hardware
//
#include <hip/hip_runtime.h>
#include <math.h>

// ---------------------------------------------------------------------------
// CViViT forward for MI455X (gfx1250, wave32, WMMA 16x16x32 f16 / f32-acc)
//   DIM=512 HEADS=8 DHEAD=64 DEPTH=4 Bv=2 T=9 TR=8 HP=16 -> NTOK=4608
//   FF_INNER=1365 (padded 1376), 2*FF=2730 (padded 2752), CODEBOOK=8192
// All GEMM dims are multiples of 32 (FF weights zero-padded into workspace),
// so the WMMA GEMM hot loop is branch-free with b128 global loads.
// ---------------------------------------------------------------------------

#define NTOK   4608
#define NFF    1365
#define NFF2   2730
#define NFFP   1376      // NFF padded to multiple of 32
#define NFF2P  2752      // NFF2 padded to multiple of 32
#define NCODE  8192

typedef __attribute__((ext_vector_type(16))) _Float16 v16h;
typedef __attribute__((ext_vector_type(8)))  float    v8f;

union AFrag { v16h v; _Float16 h[16]; };
union CFrag { v8f  v; float     f[8]; };

enum { GF_BIAS = 1, GF_ACCUM = 2 };

__device__ __forceinline__ void cvt4(AFrag& f, int o, float4 x)
{
  f.h[o + 0] = (_Float16)x.x;
  f.h[o + 1] = (_Float16)x.y;
  f.h[o + 2] = (_Float16)x.z;
  f.h[o + 3] = (_Float16)x.w;
}

// ---------------------------------------------------------------------------
// WMMA GEMM: C[M,N] (+)= A[M,K] x B[K,N] (+bias[n]).  Row-major fp32 in/out,
// f16 multiply, f32 accumulate.  REQUIRES M%32==0, N%32==0, K%32==0.
// One wave computes a 32x32 output block: 4 accumulators, 2 A-frags x
// 2 B-frags, 4 v_wmma_f32_16x16x32_f16 per K-step, all loads unconditional
// (A float4, B scalar).
// Fragment layouts per CDNA5 ISA 7.12.2 (wave32):
//   A (16x32 f16): lane L -> m=L&15, hs=L>>4; halves 0..7 -> k=8hs..8hs+7,
//                  halves 8..15 -> k=16+8hs..16+8hs+7
//   B (32x16 f16): lane L -> n=L&15, hs=L>>4; halves 0..15 -> same k as A
//   C/D (16x16 f32): lane L -> n=L&15; vgpr r -> m = r + 8*hs
// ---------------------------------------------------------------------------
__global__ void __launch_bounds__(128)
gemm_wmma_kernel(const float* __restrict__ A, const float* __restrict__ B,
                 float* __restrict__ C, int M, int N, int K,
                 const float* __restrict__ bias, int flags,
                 int tn_cnt, int tiles)
{
  int wave = threadIdx.x >> 5;
  int lane = threadIdx.x & 31;
  int tile = blockIdx.x * 4 + wave;
  if (tile >= tiles) return;
  int tm = tile / tn_cnt;
  int tn = tile - tm * tn_cnt;

  int hs = lane >> 4;
  int ml = lane & 15;

  const float* Ar0 = A + (size_t)(tm * 32 + ml)      * K + 8 * hs;
  const float* Ar1 = A + (size_t)(tm * 32 + 16 + ml) * K + 8 * hs;
  const float* Br  = B + (size_t)(8 * hs) * N + tn * 32 + ml;

  v8f acc00 = {}, acc01 = {}, acc10 = {}, acc11 = {};

  for (int k0 = 0; k0 < K; k0 += 32) {
    AFrag a0, a1, b0, b1;
    cvt4(a0, 0,  *(const float4*)(Ar0 + k0));
    cvt4(a0, 4,  *(const float4*)(Ar0 + k0 + 4));
    cvt4(a0, 8,  *(const float4*)(Ar0 + k0 + 16));
    cvt4(a0, 12, *(const float4*)(Ar0 + k0 + 20));
    cvt4(a1, 0,  *(const float4*)(Ar1 + k0));
    cvt4(a1, 4,  *(const float4*)(Ar1 + k0 + 4));
    cvt4(a1, 8,  *(const float4*)(Ar1 + k0 + 16));
    cvt4(a1, 12, *(const float4*)(Ar1 + k0 + 20));
    const float* bp = Br + (size_t)k0 * N;
#pragma unroll
    for (int s = 0; s < 16; ++s) {
      size_t kr = (size_t)(s < 8 ? s : s + 8) * N;
      b0.h[s] = (_Float16)bp[kr];
      b1.h[s] = (_Float16)bp[kr + 16];
    }
    acc00 = __builtin_amdgcn_wmma_f32_16x16x32_f16(false, a0.v, false, b0.v, (short)0, acc00, false, false);
    acc01 = __builtin_amdgcn_wmma_f32_16x16x32_f16(false, a0.v, false, b1.v, (short)0, acc01, false, false);
    acc10 = __builtin_amdgcn_wmma_f32_16x16x32_f16(false, a1.v, false, b0.v, (short)0, acc10, false, false);
    acc11 = __builtin_amdgcn_wmma_f32_16x16x32_f16(false, a1.v, false, b1.v, (short)0, acc11, false, false);
  }

  auto store_tile = [&](v8f accv, int moff, int noff) {
    CFrag c; c.v = accv;
    int n = tn * 32 + noff + ml;
#pragma unroll
    for (int r = 0; r < 8; ++r) {
      int m = tm * 32 + moff + r + 8 * hs;
      float val = c.f[r];
      if (flags & GF_BIAS) val += bias[n];
      float* cp = C + (size_t)m * N + n;
      if (flags & GF_ACCUM) val += *cp;
      *cp = val;
    }
  };
  store_tile(acc00, 0, 0);
  store_tile(acc01, 0, 16);
  store_tile(acc10, 16, 0);
  store_tile(acc11, 16, 16);
}

// ---------------------------------------------------------------------------
// Row LayerNorm: one 256-thread block per row (cols <= 512 here).
// Safe in-place (y may alias x).
// ---------------------------------------------------------------------------
__global__ void ln_kernel(const float* __restrict__ x, const float* __restrict__ g,
                          const float* __restrict__ b, float* __restrict__ y, int cols)
{
  __shared__ float red[256];
  int row = blockIdx.x, tid = threadIdx.x;
  const float* xr = x + (size_t)row * cols;
  float s = 0.f;
  for (int c = tid; c < cols; c += 256) s += xr[c];
  red[tid] = s; __syncthreads();
  for (int st = 128; st > 0; st >>= 1) { if (tid < st) red[tid] += red[tid + st]; __syncthreads(); }
  float mu = red[0] / (float)cols;
  __syncthreads();
  float s2 = 0.f;
  for (int c = tid; c < cols; c += 256) { float d = xr[c] - mu; s2 += d * d; }
  red[tid] = s2; __syncthreads();
  for (int st = 128; st > 0; st >>= 1) { if (tid < st) red[tid] += red[tid + st]; __syncthreads(); }
  float rstd = rsqrtf(red[0] / (float)cols + 1e-5f);
  __syncthreads();
  float* yr = y + (size_t)row * cols;
  for (int c = tid; c < cols; c += 256) yr[c] = (xr[c] - mu) * rstd * g[c] + b[c];
}

// ---------------------------------------------------------------------------
// Attention: one 64-thread block per (batch, head, query).  Q scaled by
// DHEAD^-0.5; optional CPB bias (8,Nseq,Nseq); optional causal+ALiBi (slopes
// 2^-(h+1) for HEADS=8).  KV layout: cols[0:512]=k, [512:1024]=v.
// ---------------------------------------------------------------------------
__global__ void attn_kernel(const float* __restrict__ Q, const float* __restrict__ KV,
                            float* __restrict__ O, int Nseq,
                            const float* __restrict__ bias, int causal)
{
  __shared__ float sim[256];
  __shared__ float qs[64];
  __shared__ float red[64];
  int gid = blockIdx.x;
  int i  = gid % Nseq;
  int h  = (gid / Nseq) & 7;
  int bb = gid / (Nseq * 8);
  int tid = threadIdx.x;

  qs[tid] = Q[((size_t)(bb * Nseq + i) * 512) + h * 64 + tid] * 0.125f;
  __syncthreads();

  float slope = exp2f(-(float)(h + 1));
  for (int j = tid; j < Nseq; j += 64) {
    const float* krow = KV + ((size_t)(bb * Nseq + j) * 1024) + h * 64;
    float s = 0.f;
    for (int d = 0; d < 64; ++d) s += qs[d] * krow[d];
    if (bias) s += bias[((size_t)h * Nseq + i) * Nseq + j];
    if (causal) {
      s -= slope * fabsf((float)(i - j));
      if (j > i) s = -3.0e38f;
    }
    sim[j] = s;
  }
  __syncthreads();

  float m = -3.4e38f;
  for (int j = tid; j < Nseq; j += 64) m = fmaxf(m, sim[j]);
  red[tid] = m; __syncthreads();
  for (int st = 32; st > 0; st >>= 1) { if (tid < st) red[tid] = fmaxf(red[tid], red[tid + st]); __syncthreads(); }
  m = red[0]; __syncthreads();

  float ssum = 0.f;
  for (int j = tid; j < Nseq; j += 64) { float e = expf(sim[j] - m); sim[j] = e; ssum += e; }
  red[tid] = ssum; __syncthreads();
  for (int st = 32; st > 0; st >>= 1) { if (tid < st) red[tid] += red[tid + st]; __syncthreads(); }
  float inv = 1.0f / red[0];
  __syncthreads();

  const float* vbase = KV + 512 + h * 64 + tid;
  float acc = 0.f;
  for (int j = 0; j < Nseq; ++j) acc += sim[j] * vbase[(size_t)(bb * Nseq + j) * 1024];
  O[((size_t)(bb * Nseq + i) * 512) + h * 64 + tid] = acc * inv;
}

// GeGLU into padded FFA (stride NFFP): cols >= NFF are zero-filled so the
// following K=NFFP GEMM needs no bounds checks.
__global__ void geglu_kernel(const float* __restrict__ h, float* __restrict__ out)
{
  int idx = blockIdx.x * 256 + threadIdx.x;
  if (idx >= NTOK * NFFP) return;
  int row = idx / NFFP, col = idx - row * NFFP;
  float r = 0.f;
  if (col < NFF) {
    float a = h[(size_t)row * NFF2P + col];
    float g = h[(size_t)row * NFF2P + NFF + col];
    r = 0.5f * g * (1.f + erff(g * 0.70710678118654752f)) * a;
  }
  out[idx] = r;
}

// Zero-pad FF weights into aligned workspace copies.
__global__ void padw1_kernel(const float* __restrict__ w1, float* __restrict__ dst)
{ // (512, 2730) -> (512, 2752)
  int idx = blockIdx.x * 256 + threadIdx.x;
  if (idx >= 512 * NFF2P) return;
  int row = idx / NFF2P, col = idx - row * NFF2P;
  dst[idx] = (col < NFF2) ? w1[(size_t)row * NFF2 + col] : 0.f;
}
__global__ void padw2_kernel(const float* __restrict__ w2, float* __restrict__ dst)
{ // (1365, 512) -> (1376, 512)
  int idx = blockIdx.x * 256 + threadIdx.x;
  if (idx >= NFFP * 512) return;
  int row = idx >> 9;
  dst[idx] = (row < NFF) ? w2[idx] : 0.f;
}

// ---------------------------------------------------------------------------
// Fused CPB relative-position MLP: (2 -> 512 leaky -> 512 leaky -> 8) per
// (i,j) pair; bias stored (8, 256, 256).
// ---------------------------------------------------------------------------
__global__ void cpb_kernel(const float* __restrict__ w0, const float* __restrict__ b0,
                           const float* __restrict__ w1, const float* __restrict__ b1,
                           const float* __restrict__ w2, const float* __restrict__ b2,
                           float* __restrict__ bias)
{
  __shared__ float h0[512];
  __shared__ float h1[512];
  int pij = blockIdx.x;             // 0..65535
  int i = pij >> 8, j = pij & 255;
  int tid = threadIdx.x;
  float dy = (float)(i >> 4) - (float)(j >> 4);
  float dx = (float)(i & 15) - (float)(j & 15);
  float sy = (dy > 0.f) ? 1.f : ((dy < 0.f) ? -1.f : 0.f);
  float sx = (dx > 0.f) ? 1.f : ((dx < 0.f) ? -1.f : 0.f);
  float r0 = sy * logf(fabsf(dy) + 1.f);
  float r1 = sx * logf(fabsf(dx) + 1.f);
  for (int d = tid; d < 512; d += 128) {
    float v = r0 * w0[d] + r1 * w0[512 + d] + b0[d];
    h0[d] = (v >= 0.f) ? v : 0.1f * v;
  }
  __syncthreads();
  for (int d = tid; d < 512; d += 128) {
    float v = b1[d];
    for (int k = 0; k < 512; ++k) v += h0[k] * w1[k * 512 + d];
    h1[d] = (v >= 0.f) ? v : 0.1f * v;
  }
  __syncthreads();
  if (tid < 8) {
    float v = b2[tid];
    for (int k = 0; k < 512; ++k) v += h1[k] * w2[k * 8 + tid];
    bias[tid * 65536 + pij] = v;
  }
}

// Codebook row norms (8192 x 512)
__global__ void cbnorm_kernel(const float* __restrict__ cb, float* __restrict__ norms)
{
  __shared__ float red[256];
  int row = blockIdx.x, tid = threadIdx.x;
  const float* cr = cb + (size_t)row * 512;
  float s = 0.f;
  for (int c = tid; c < 512; c += 256) { float v = cr[c]; s += v * v; }
  red[tid] = s; __syncthreads();
  for (int st = 128; st > 0; st >>= 1) { if (tid < st) red[tid] += red[tid + st]; __syncthreads(); }
  if (tid == 0) norms[row] = fmaxf(sqrtf(red[0]), 1e-12f);
}

// VQ: per token row, argmax_j <x, cb_j>/||cb_j|| (scale of x doesn't change
// argmax), then replace row with the normalized codebook entry.
__global__ void vq_kernel(float* __restrict__ X, const float* __restrict__ cb,
                          const float* __restrict__ cbn)
{
  __shared__ float xrow[512];
  __shared__ float bv[256];
  __shared__ int   bi[256];
  int row = blockIdx.x, tid = threadIdx.x;
  float* xr = X + (size_t)row * 512;
  for (int c = tid; c < 512; c += 256) xrow[c] = xr[c];
  __syncthreads();
  float best = -3.4e38f; int besti = 0;
  for (int j = tid; j < NCODE; j += 256) {
    const float* cr = cb + (size_t)j * 512;
    float d = 0.f;
    for (int k = 0; k < 512; ++k) d += xrow[k] * cr[k];
    d /= cbn[j];
    if (d > best) { best = d; besti = j; }
  }
  bv[tid] = best; bi[tid] = besti; __syncthreads();
  for (int st = 128; st > 0; st >>= 1) {
    if (tid < st) {
      float v2 = bv[tid + st]; int i2 = bi[tid + st];
      if (v2 > bv[tid] || (v2 == bv[tid] && i2 < bi[tid])) { bv[tid] = v2; bi[tid] = i2; }
    }
    __syncthreads();
  }
  int w = bi[0];
  float inv = 1.f / cbn[w];
  const float* cr = cb + (size_t)w * 512;
  for (int c = tid; c < 512; c += 256) xr[c] = cr[c] * inv;
}

// --------------------------- layout shuffles -------------------------------
// video (2,3,17,128,128); spatial token row rS=(b*9+t)*256+s; temporal
// rT=(b*256+s)*9+t with s = hy*16+hx.

__global__ void gather1_kernel(const float* __restrict__ video, float* __restrict__ G1)
{ // (Bv*256) x 192 : first frame patches, col=(c,p1,p2)
  int idx = blockIdx.x * 256 + threadIdx.x;
  if (idx >= 512 * 192) return;
  int row = idx / 192, col = idx - row * 192;
  int b = row >> 8, s = row & 255, hy = s >> 4, hx = s & 15;
  int c = col / 64, r = col & 63, p1 = r >> 3, p2 = r & 7;
  G1[idx] = video[((size_t)(b * 3 + c) * 17 + 0) * 16384 + (hy * 8 + p1) * 128 + hx * 8 + p2];
}

__global__ void gatherR_kernel(const float* __restrict__ video, float* __restrict__ GR)
{ // (Bv*8*256) x 384 : rest frames, col=(c,pt,p1,p2), frame=1+2*tr+pt
  int idx = blockIdx.x * 256 + threadIdx.x;
  if (idx >= 4096 * 384) return;
  int row = idx / 384, col = idx - row * 384;
  int b = row >> 11, rem = row & 2047, tr = rem >> 8, s = rem & 255;
  int hy = s >> 4, hx = s & 15;
  int c = col / 128, pt = (col >> 6) & 1, p1 = (col >> 3) & 7, p2 = col & 7;
  int frame = 1 + 2 * tr + pt;
  GR[idx] = video[((size_t)(b * 3 + c) * 17 + frame) * 16384 + (hy * 8 + p1) * 128 + hx * 8 + p2];
}

__global__ void copy_t0_in_kernel(const float* __restrict__ TP1, float* __restrict__ X)
{
  int idx = blockIdx.x * 256 + threadIdx.x;
  if (idx >= 512 * 512) return;
  int row = idx >> 9, col = idx & 511;
  int b = row >> 8, s = row & 255;
  X[(size_t)(b * 2304 + s) * 512 + col] = TP1[idx];
}

__global__ void copy_rest_in_kernel(const float* __restrict__ TPR, float* __restrict__ X)
{
  int idx = blockIdx.x * 256 + threadIdx.x;
  if (idx >= 4096 * 512) return;
  int row = idx >> 9, col = idx & 511;
  int b = row >> 11, rem = row & 2047, tr = rem >> 8, s = rem & 255;
  X[(size_t)((b * 9 + 1 + tr) * 256 + s) * 512 + col] = TPR[idx];
}

__global__ void reorder_s2t_kernel(const float* __restrict__ src, float* __restrict__ dst)
{
  int idx = blockIdx.x * 256 + threadIdx.x;
  if (idx >= NTOK * 512) return;
  int row = idx >> 9, col = idx & 511;
  int b = row / 2304, rem = row - b * 2304, t = rem >> 8, s = rem & 255;
  dst[(size_t)((b * 256 + s) * 9 + t) * 512 + col] = src[idx];
}

__global__ void reorder_t2s_kernel(const float* __restrict__ src, float* __restrict__ dst)
{
  int idx = blockIdx.x * 256 + threadIdx.x;
  if (idx >= NTOK * 512) return;
  int row = idx >> 9, col = idx & 511;
  int b = row / 2304, rem = row - b * 2304, s = rem / 9, t = rem - s * 9;
  dst[(size_t)((b * 9 + t) * 256 + s) * 512 + col] = src[idx];
}

__global__ void pack_t0_kernel(const float* __restrict__ X, float* __restrict__ A1)
{
  int idx = blockIdx.x * 256 + threadIdx.x;
  if (idx >= 512 * 512) return;
  int row = idx >> 9, col = idx & 511;
  int b = row >> 8, s = row & 255;
  A1[idx] = X[(size_t)(b * 2304 + s) * 512 + col];
}

__global__ void pack_rest_kernel(const float* __restrict__ X, float* __restrict__ AR)
{
  int idx = blockIdx.x * 256 + threadIdx.x;
  if (idx >= 4096 * 512) return;
  int row = idx >> 9, col = idx & 511;
  int b = row >> 11, rem = row & 2047, tr = rem >> 8, s = rem & 255;
  AR[idx] = X[(size_t)((b * 9 + 1 + tr) * 256 + s) * 512 + col];
}

__global__ void scatter_f1_kernel(const float* __restrict__ F1, float* __restrict__ out)
{
  int idx = blockIdx.x * 256 + threadIdx.x;
  if (idx >= 512 * 192) return;
  int row = idx / 192, col = idx - row * 192;
  int b = row >> 8, s = row & 255, hy = s >> 4, hx = s & 15;
  int c = col / 64, r = col & 63, p1 = r >> 3, p2 = r & 7;
  out[((size_t)(b * 3 + c) * 17 + 0) * 16384 + (hy * 8 + p1) * 128 + hx * 8 + p2] = F1[idx];
}

__global__ void scatter_fr_kernel(const float* __restrict__ FR, float* __restrict__ out)
{
  int idx = blockIdx.x * 256 + threadIdx.x;
  if (idx >= 4096 * 384) return;
  int row = idx / 384, col = idx - row * 384;
  int b = row >> 11, rem = row & 2047, tr = rem >> 8, s = rem & 255;
  int hy = s >> 4, hx = s & 15;
  int c = col / 128, pt = (col >> 6) & 1, p1 = (col >> 3) & 7, p2 = col & 7;
  int frame = 1 + 2 * tr + pt;
  out[((size_t)(b * 3 + c) * 17 + frame) * 16384 + (hy * 8 + p1) * 128 + hx * 8 + p2] = FR[idx];
}

// --------------------------- host-side helpers -----------------------------

static inline void launch_gemm(hipStream_t s, const float* A, const float* B, float* C,
                               int M, int N, int K, const float* bias, int flags)
{
  int tm = M / 32, tn = N / 32;       // all dims multiples of 32 by design
  int tiles = tm * tn;
  int blocks = (tiles + 3) / 4;
  gemm_wmma_kernel<<<blocks, 128, 0, s>>>(A, B, C, M, N, K, bias, flags, tn, tiles);
}

static void run_transformer(hipStream_t s, float* X, int it, int Bn, int Nseq,
                            const float* bias, int causal,
                            const float* ln1g, const float* ln1b,
                            const float* wq, const float* wkv, const float* wo,
                            const float* ffg, const float* ffb,
                            const float* w1, const float* w2,
                            const float* outg, const float* outb,
                            float* XLN, float* Qb, float* KVb, float* Ob,
                            float* FFHb, float* FFAb, float* W1P, float* W2P)
{
  for (int l = 0; l < 4; ++l) {
    size_t li = (size_t)(it * 4 + l);
    const float* g   = ln1g + li * 512;
    const float* b   = ln1b + li * 512;
    const float* wql = wq   + li * 512 * 512;
    const float* wkl = wkv  + li * 512 * 1024;
    const float* wol = wo   + li * 512 * 512;
    const float* fg  = ffg  + li * 512;
    const float* fb  = ffb  + li * 512;
    const float* w1l = w1   + li * 512 * NFF2;
    const float* w2l = w2   + li * NFF * 512;

    ln_kernel<<<NTOK, 256, 0, s>>>(X, g, b, XLN, 512);
    launch_gemm(s, XLN, wql, Qb,  NTOK, 512,  512, nullptr, 0);
    launch_gemm(s, X,   wkl, KVb, NTOK, 1024, 512, nullptr, 0);
    attn_kernel<<<Bn * 8 * Nseq, 64, 0, s>>>(Qb, KVb, Ob, Nseq, bias, causal);
    launch_gemm(s, Ob, wol, X, NTOK, 512, 512, nullptr, GF_ACCUM);

    ln_kernel<<<NTOK, 256, 0, s>>>(X, fg, fb, XLN, 512);
    padw1_kernel<<<(512 * NFF2P + 255) / 256, 256, 0, s>>>(w1l, W1P);
    launch_gemm(s, XLN, W1P, FFHb, NTOK, NFF2P, 512, nullptr, 0);
    geglu_kernel<<<(NTOK * NFFP + 255) / 256, 256, 0, s>>>(FFHb, FFAb);
    padw2_kernel<<<(NFFP * 512 + 255) / 256, 256, 0, s>>>(w2l, W2P);
    launch_gemm(s, FFAb, W2P, X, NTOK, 512, NFFP, nullptr, GF_ACCUM);
  }
  ln_kernel<<<NTOK, 256, 0, s>>>(X, outg + (size_t)it * 512, outb + (size_t)it * 512, X, 512);
}

extern "C" void kernel_launch(void* const* d_in, const int* in_sizes, int n_in,
                              void* d_out, int out_size, void* d_ws, size_t ws_size,
                              hipStream_t stream)
{
  (void)in_sizes; (void)n_in; (void)out_size; (void)ws_size;
  const float* video    = (const float*)d_in[0];
  const float* pe1_ln_g = (const float*)d_in[1];
  const float* pe1_ln_b = (const float*)d_in[2];
  const float* pe1_w    = (const float*)d_in[3];
  const float* pe1_b    = (const float*)d_in[4];
  const float* pe1_l2g  = (const float*)d_in[5];
  const float* pe1_l2b  = (const float*)d_in[6];
  const float* pe_ln_g  = (const float*)d_in[7];
  const float* pe_ln_b  = (const float*)d_in[8];
  const float* pe_w     = (const float*)d_in[9];
  const float* pe_b     = (const float*)d_in[10];
  const float* pe_l2g   = (const float*)d_in[11];
  const float* pe_l2b   = (const float*)d_in[12];
  const float* cpb_w0   = (const float*)d_in[13];
  const float* cpb_b0   = (const float*)d_in[14];
  const float* cpb_w1   = (const float*)d_in[15];
  const float* cpb_b1   = (const float*)d_in[16];
  const float* cpb_w2   = (const float*)d_in[17];
  const float* cpb_b2   = (const float*)d_in[18];
  const float* tf_ln1_g = (const float*)d_in[19];
  const float* tf_ln1_b = (const float*)d_in[20];
  const float* tf_wq    = (const float*)d_in[21];
  const float* tf_wkv   = (const float*)d_in[22];
  const float* tf_wo    = (const float*)d_in[23];
  const float* tf_ffg   = (const float*)d_in[24];
  const float* tf_ffb   = (const float*)d_in[25];
  const float* tf_w1    = (const float*)d_in[26];
  const float* tf_w2    = (const float*)d_in[27];
  const float* tf_outg  = (const float*)d_in[28];
  const float* tf_outb  = (const float*)d_in[29];
  const float* codebook = (const float*)d_in[30];
  const float* px1_w    = (const float*)d_in[31];
  const float* px1_b    = (const float*)d_in[32];
  const float* px_w     = (const float*)d_in[33];
  const float* px_b     = (const float*)d_in[34];
  float* out = (float*)d_out;

  // ---- workspace layout (floats) ----
  float* ws = (float*)d_ws;
  size_t off = 0;
  float* XA   = ws + off; off += (size_t)NTOK * 512;
  float* XB   = ws + off; off += (size_t)NTOK * 512;
  float* XLN  = ws + off; off += (size_t)NTOK * 512;
  float* Qb   = ws + off; off += (size_t)NTOK * 512;
  float* KVb  = ws + off; off += (size_t)NTOK * 1024;
  float* Ob   = ws + off; off += (size_t)NTOK * 512;
  float* FFHb = ws + off; off += (size_t)NTOK * NFF2P;   // also pe/pixel scratch
  float* FFAb = ws + off; off += (size_t)NTOK * NFFP;
  float* W1P  = ws + off; off += (size_t)512 * NFF2P;
  float* W2P  = ws + off; off += (size_t)NFFP * 512;
  float* BIAS = ws + off; off += (size_t)8 * 256 * 256;
  float* CBN  = ws + off; off += (size_t)NCODE;

  // pe-phase overlay inside FFHb (used before transformers only)
  float* G1  = FFHb;
  float* G1N = FFHb + 98304;
  float* TP1 = FFHb + 196608;
  float* GR  = FFHb + 458752;
  float* GRN = FFHb + 2031616;
  float* TPR = FFHb + 3604480;
  // pixel-phase overlay inside FFHb (used after transformers only)
  float* A1 = FFHb;
  float* AR = FFHb + 262144;
  float* F1 = FFHb + 2359296;
  float* FR = FFHb + 2457600;

  // 1) relative position bias (8,256,256)
  cpb_kernel<<<65536, 128, 0, stream>>>(cpb_w0, cpb_b0, cpb_w1, cpb_b1, cpb_w2, cpb_b2, BIAS);

  // 2) patch embeddings -> XA (spatial layout)
  gather1_kernel<<<(512 * 192 + 255) / 256, 256, 0, stream>>>(video, G1);
  ln_kernel<<<512, 256, 0, stream>>>(G1, pe1_ln_g, pe1_ln_b, G1N, 192);
  launch_gemm(stream, G1N, pe1_w, TP1, 512, 512, 192, pe1_b, GF_BIAS);
  ln_kernel<<<512, 256, 0, stream>>>(TP1, pe1_l2g, pe1_l2b, TP1, 512);
  copy_t0_in_kernel<<<(512 * 512 + 255) / 256, 256, 0, stream>>>(TP1, XA);

  gatherR_kernel<<<(4096 * 384 + 255) / 256, 256, 0, stream>>>(video, GR);
  ln_kernel<<<4096, 256, 0, stream>>>(GR, pe_ln_g, pe_ln_b, GRN, 384);
  launch_gemm(stream, GRN, pe_w, TPR, 4096, 512, 384, pe_b, GF_BIAS);
  ln_kernel<<<4096, 256, 0, stream>>>(TPR, pe_l2g, pe_l2b, TPR, 512);
  copy_rest_in_kernel<<<(4096 * 512 + 255) / 256, 256, 0, stream>>>(TPR, XA);

  // 3) spatial transformer 0 (bias)
  run_transformer(stream, XA, 0, 18, 256, BIAS, 0, tf_ln1_g, tf_ln1_b, tf_wq, tf_wkv,
                  tf_wo, tf_ffg, tf_ffb, tf_w1, tf_w2, tf_outg, tf_outb,
                  XLN, Qb, KVb, Ob, FFHb, FFAb, W1P, W2P);

  // 4) -> temporal layout; transformer 1 (causal+alibi)
  reorder_s2t_kernel<<<(NTOK * 512 + 255) / 256, 256, 0, stream>>>(XA, XB);
  run_transformer(stream, XB, 1, 512, 9, nullptr, 1, tf_ln1_g, tf_ln1_b, tf_wq, tf_wkv,
                  tf_wo, tf_ffg, tf_ffb, tf_w1, tf_w2, tf_outg, tf_outb,
                  XLN, Qb, KVb, Ob, FFHb, FFAb, W1P, W2P);

  // 5) vector quantize (per-token; layout independent)
  cbnorm_kernel<<<NCODE, 256, 0, stream>>>(codebook, CBN);
  vq_kernel<<<NTOK, 256, 0, stream>>>(XB, codebook, CBN);

  // 6) transformer 2 (temporal causal), back to spatial, transformer 3 (bias)
  run_transformer(stream, XB, 2, 512, 9, nullptr, 1, tf_ln1_g, tf_ln1_b, tf_wq, tf_wkv,
                  tf_wo, tf_ffg, tf_ffb, tf_w1, tf_w2, tf_outg, tf_outb,
                  XLN, Qb, KVb, Ob, FFHb, FFAb, W1P, W2P);
  reorder_t2s_kernel<<<(NTOK * 512 + 255) / 256, 256, 0, stream>>>(XB, XA);
  run_transformer(stream, XA, 3, 18, 256, BIAS, 0, tf_ln1_g, tf_ln1_b, tf_wq, tf_wkv,
                  tf_wo, tf_ffg, tf_ffb, tf_w1, tf_w2, tf_outg, tf_outb,
                  XLN, Qb, KVb, Ob, FFHb, FFAb, W1P, W2P);

  // 7) pixel heads -> output video
  pack_t0_kernel<<<(512 * 512 + 255) / 256, 256, 0, stream>>>(XA, A1);
  pack_rest_kernel<<<(4096 * 512 + 255) / 256, 256, 0, stream>>>(XA, AR);
  launch_gemm(stream, A1, px1_w, F1, 512, 192, 512, px1_b, GF_BIAS);
  launch_gemm(stream, AR, px_w, FR, 4096, 384, 512, px_b, GF_BIAS);
  scatter_f1_kernel<<<(512 * 192 + 255) / 256, 256, 0, stream>>>(F1, out);
  scatter_fr_kernel<<<(4096 * 384 + 255) / 256, 256, 0, stream>>>(FR, out);
}
